// Attention_74732430950409
// MI455X (gfx1250) — matrix operand order repaired and checked by reading, not compiled, not run on hardware
//
#include <hip/hip_runtime.h>
#include <math.h>

// ---------------------------------------------------------------------------
// Types
// ---------------------------------------------------------------------------
typedef __bf16        bf16x16 __attribute__((ext_vector_type(16)));
typedef float         v8f     __attribute__((ext_vector_type(8)));
typedef unsigned int  u32x4   __attribute__((ext_vector_type(4)));
typedef int           i32x4   __attribute__((ext_vector_type(4)));
typedef int           i32x8   __attribute__((ext_vector_type(8)));

struct Pack32 { u32x4 lo, hi; };   // 32 bytes == one bf16x16 fragment

static __device__ __forceinline__ bf16x16 make_frag(const unsigned short* p0,
                                                    const unsigned short* p1) {
    Pack32 pk;
    pk.lo = *(const u32x4*)p0;
    pk.hi = *(const u32x4*)p1;
    return __builtin_bit_cast(bf16x16, pk);
}

static __device__ __forceinline__ unsigned short f2bf(float f) {
    unsigned int u = __float_as_uint(f);
    u = u + 0x7FFFu + ((u >> 16) & 1u);      // round-to-nearest-even
    return (unsigned short)(u >> 16);
}
static __device__ __forceinline__ float bf2f(unsigned short h) {
    return __uint_as_float(((unsigned int)h) << 16);
}

// ---------------------------------------------------------------------------
// TDM: async 2D tile load Global -> LDS, tracked by TENSORcnt.
// Tile: tile_rows x 32 bf16 elements, tensor row stride = K elements.
// LDS rows padded by the DMA engine: pad_interval=16 DWORDs (one 64B row),
// pad_amount=4 DWORDs (16B)  ->  80B LDS row stride (= 40 bf16 elems).
// ---------------------------------------------------------------------------
#define GEMM_TK    32
#define GEMM_LDSTR 40

static __device__ __forceinline__ void tdm_load_tile(unsigned lds_off,
                                                     const unsigned short* gptr,
                                                     int rows_total, int K,
                                                     int tile_rows) {
    unsigned long long ga = (unsigned long long)(const void*)gptr;
    u32x4 g0;
    g0[0] = 1u;                                        // count=1, no gather
    g0[1] = lds_off;                                   // lds_addr [63:32]
    g0[2] = (unsigned)(ga & 0xffffffffu);              // global_addr lo
    g0[3] = (unsigned)((ga >> 32) & 0x1ffffffu)        // global_addr [56:32]
          | (2u << 30);                                // type=2 ("image")
    i32x8 g1;
    g1[0] = (int)((1u << 16)      // data_size = 2 bytes
                | (1u << 20)      // pad_enable
                | (3u << 22)      // pad_interval: 16 DWORDs
                | (3u << 25));    // pad_amount:   4 DWORDs
    g1[1] = (int)(((unsigned)K & 0xffffu) << 16);                      // tensor_dim0 lo16
    g1[2] = (int)((((unsigned)K >> 16) & 0xffffu)
                | (((unsigned)rows_total & 0xffffu) << 16));           // dim0 hi / dim1 lo
    g1[3] = (int)((((unsigned)rows_total >> 16) & 0xffffu)
                | ((unsigned)GEMM_TK << 16));                          // dim1 hi / tile_dim0
    g1[4] = (int)((unsigned)tile_rows & 0xffffu);                      // tile_dim1 (tile_dim2=0)
    g1[5] = K;                                                         // tensor_dim0_stride lo32
    g1[6] = 0;
    g1[7] = 0;
    i32x4 z4 = {0, 0, 0, 0};
#if __clang_major__ >= 23
    i32x8 z8 = {0, 0, 0, 0, 0, 0, 0, 0};
    __builtin_amdgcn_tensor_load_to_lds(g0, g1, z4, z4, z8, 0);
#else
    __builtin_amdgcn_tensor_load_to_lds(g0, g1, z4, z4, 0);
#endif
}

// ---------------------------------------------------------------------------
// RoPE + fp32 -> bf16 (one block per sequence row)
// ---------------------------------------------------------------------------
__global__ void __launch_bounds__(256)
rope_to_bf16(const float* __restrict__ x, unsigned short* __restrict__ y,
             int S, int D) {
    const int s = blockIdx.x;
    const int half = D >> 1;                 // 1024
    const float* row = x + (size_t)s * D;
    unsigned short* o = y + (size_t)s * D;
    const float kLogTheta = 9.210340371976184f;   // ln(10000)
    const float invHalf = 1.0f / (float)half;
    for (int c = threadIdx.x; c < half; c += blockDim.x) {
        float freq = __expf(-(float)c * invHalf * kLogTheta);
        float ang  = (float)s * freq;
        float cs, sn;
        __sincosf(ang, &sn, &cs);
        float a = row[c], b = row[c + half];
        o[c]        = f2bf(a * cs - b * sn);
        o[c + half] = f2bf(b * cs + a * sn);
    }
}

// ---------------------------------------------------------------------------
// fp32 -> bf16 elementwise
// ---------------------------------------------------------------------------
__global__ void __launch_bounds__(256)
convert_f32_bf16(const float* __restrict__ src, unsigned short* __restrict__ dst,
                 int n) {
    int i = blockIdx.x * 256 + threadIdx.x;
    if (i < n) dst[i] = f2bf(src[i]);
}

// ---------------------------------------------------------------------------
// fp32 -> bf16 with transpose: dst[c*R + r] = src[r*C + c]   (32x32 LDS tiles)
// ---------------------------------------------------------------------------
__global__ void __launch_bounds__(256)
transpose_f32_bf16(const float* __restrict__ src, unsigned short* __restrict__ dst,
                   int R, int C) {
    __shared__ float tile[32][33];
    const int bc = blockIdx.x * 32;
    const int br = blockIdx.y * 32;
    const int tx = threadIdx.x & 31;
    const int ty = threadIdx.x >> 5;         // 0..7
    for (int i = 0; i < 32; i += 8)
        tile[ty + i][tx] = src[(size_t)(br + ty + i) * C + bc + tx];
    __syncthreads();
    for (int i = 0; i < 32; i += 8)
        dst[(size_t)(bc + ty + i) * R + br + tx] = f2bf(tile[tx][ty + i]);
}

// ---------------------------------------------------------------------------
// bf16 NT GEMM:  C[M,N] = A[M,K] * B[N,K]^T   (fp32 accumulate, bf16 out)
// Block: 256 threads (8 waves), 128x256 C tile, K-step 32.
// Wave w: 64x64 sub-tile -> 4x4 grid of 16x16 WMMA accumulators.
// Tiles staged by TDM (tensor_load_to_lds) with double-buffered LDS:
// DMA of stage k+1 overlaps WMMA of stage k, synchronized via TENSORcnt.
// ---------------------------------------------------------------------------
__global__ void __launch_bounds__(256)
gemm_nt_bf16(const unsigned short* __restrict__ A,
             const unsigned short* __restrict__ B,
             unsigned short* __restrict__ C,
             int M, int N, int K) {
    __shared__ unsigned short As[2][128 * GEMM_LDSTR];   // 2 x 10240 B
    __shared__ unsigned short Bs[2][256 * GEMM_LDSTR];   // 2 x 20480 B

    const int t     = threadIdx.x;
    const int lane  = t & 31;
    const int wave0 = (t < 32);
    const int wave  = t >> 5;
    const int m0    = blockIdx.y * 128;
    const int n0    = blockIdx.x * 256;
    const int wm    = (wave & 1) * 64;       // 2 waves along M
    const int wn    = (wave >> 1) * 64;      // 4 waves along N
    const int lrow  = lane & 15;
    const int lhalf = lane >> 4;             // 0 or 1

    v8f acc[4][4];
    for (int i = 0; i < 4; ++i)
        for (int j = 0; j < 4; ++j)
            acc[i][j] = {};

    const int nk = K / GEMM_TK;

    // prologue: stage 0 into buffer 0
    if (wave0) {
        tdm_load_tile((unsigned)(unsigned long long)(void*)&As[0][0],
                      &A[(size_t)m0 * K], M, K, 128);
        tdm_load_tile((unsigned)(unsigned long long)(void*)&Bs[0][0],
                      &B[(size_t)n0 * K], N, K, 256);
    }

    for (int ks = 0; ks < nk; ++ks) {
        const int cur = ks & 1;
        if (wave0) {
            if (ks + 1 < nk) {
                const int nxt = cur ^ 1;
                const int k1  = (ks + 1) * GEMM_TK;
                tdm_load_tile((unsigned)(unsigned long long)(void*)&As[nxt][0],
                              &A[(size_t)m0 * K + k1], M, K, 128);
                tdm_load_tile((unsigned)(unsigned long long)(void*)&Bs[nxt][0],
                              &B[(size_t)n0 * K + k1], N, K, 256);
                __builtin_amdgcn_s_wait_tensorcnt(2);   // stage ks complete
            } else {
                __builtin_amdgcn_s_wait_tensorcnt(0);
            }
        }
        __syncthreads();

        bf16x16 afr[4], bfr[4];
        for (int tm = 0; tm < 4; ++tm) {
            const unsigned short* p =
                &As[cur][(wm + tm * 16 + lrow) * GEMM_LDSTR + lhalf * 8];
            afr[tm] = make_frag(p, p + 16);
        }
        for (int tn = 0; tn < 4; ++tn) {
            const unsigned short* p =
                &Bs[cur][(wn + tn * 16 + lrow) * GEMM_LDSTR + lhalf * 8];
            bfr[tn] = make_frag(p, p + 16);
        }

        for (int tm = 0; tm < 4; ++tm)
            for (int tn = 0; tn < 4; ++tn)
                acc[tm][tn] = __builtin_amdgcn_wmma_f32_16x16x32_bf16(
                    false, afr[tm], false, bfr[tn],
                    (short)0, acc[tm][tn], false, false);
        __syncthreads();   // all waves done reading buf 'cur' before its reuse
    }

    // store: VGPR r -> M = r + lhalf*8, N = lrow (per ISA C-layout)
    for (int tm = 0; tm < 4; ++tm)
        for (int tn = 0; tn < 4; ++tn)
            for (int r = 0; r < 8; ++r) {
                int mm = m0 + wm + tm * 16 + r + lhalf * 8;
                int nn = n0 + wn + tn * 16 + lrow;
                C[(size_t)mm * N + nn] = f2bf(acc[tm][tn][r]);
            }
}

// ---------------------------------------------------------------------------
// fp32 matvec: y[o] = dot(W[o,:], x) * scale (+ residual[o])
// ---------------------------------------------------------------------------
__global__ void __launch_bounds__(256)
matvec_f32(const float* __restrict__ W, const float* __restrict__ x,
           const float* __restrict__ residual, float* __restrict__ y,
           int O, int I, float scale) {
    const int row  = blockIdx.x * 8 + (threadIdx.x >> 5);
    const int lane = threadIdx.x & 31;
    if (row >= O) return;
    const float* w = W + (size_t)row * I;
    float acc = 0.0f;
    for (int i = lane; i < I; i += 32) acc += w[i] * x[i];
    for (int off = 16; off > 0; off >>= 1) acc += __shfl_xor(acc, off, 32);
    if (lane == 0) {
        float r = residual ? residual[row] : 0.0f;
        y[row] = acc * scale + r;
    }
}

// ---------------------------------------------------------------------------
// Attention core: one block per head. Logits kept in LDS, block softmax,
// then streaming weighted sum over vh.  Bandwidth-bound -> fp32 VALU.
// ---------------------------------------------------------------------------
__global__ void __launch_bounds__(256)
attn_softmax_wsum(const float* __restrict__ qh,
                  const unsigned short* __restrict__ kh,
                  const unsigned short* __restrict__ vh,
                  float* __restrict__ attn, int S) {
    __shared__ float logits[8192];
    __shared__ float qs[128];
    __shared__ float red[256];

    const int h = blockIdx.x;
    const int t = threadIdx.x;
    const float rscale = 0.08838834764831845f;   // 1/sqrt(128)

    if (t < 128) qs[t] = qh[h * 128 + t];
    __syncthreads();

    float lmax = -1e30f;
    for (int s = t; s < S; s += 256) {
        const unsigned short* krow = kh + (size_t)s * 2048 + h * 128;
        float d = 0.0f;
        for (int c = 0; c < 128; c += 8) {
            u32x4 pk = *(const u32x4*)&krow[c];
            for (int j = 0; j < 4; ++j) {
                unsigned int w = pk[j];
                d += qs[c + 2 * j]     * bf2f((unsigned short)(w & 0xffffu));
                d += qs[c + 2 * j + 1] * bf2f((unsigned short)(w >> 16));
            }
        }
        d *= rscale;
        logits[s] = d;
        lmax = fmaxf(lmax, d);
    }
    red[t] = lmax;
    __syncthreads();
    for (int o = 128; o > 0; o >>= 1) {
        if (t < o) red[t] = fmaxf(red[t], red[t + o]);
        __syncthreads();
    }
    const float m = red[0];
    __syncthreads();

    float lsum = 0.0f;
    for (int s = t; s < S; s += 256) {
        float e = __expf(logits[s] - m);
        logits[s] = e;
        lsum += e;
    }
    red[t] = lsum;
    __syncthreads();
    for (int o = 128; o > 0; o >>= 1) {
        if (t < o) red[t] += red[t + o];
        __syncthreads();
    }
    const float inv = 1.0f / red[0];
    __syncthreads();

    const int d    = t & 127;
    const int half = t >> 7;
    const int sBeg = half * (S >> 1);
    const int sEnd = sBeg + (S >> 1);
    float acc = 0.0f;
    for (int s = sBeg; s < sEnd; ++s)
        acc += logits[s] * bf2f(vh[(size_t)s * 2048 + h * 128 + d]);
    red[t] = acc;
    __syncthreads();
    if (t < 128) attn[h * 128 + t] = (red[t] + red[t + 128]) * inv;
}

// ---------------------------------------------------------------------------
// Host launcher
// ---------------------------------------------------------------------------
extern "C" void kernel_launch(void* const* d_in, const int* in_sizes, int n_in,
                              void* d_out, int out_size, void* d_ws, size_t ws_size,
                              hipStream_t stream) {
    (void)in_sizes; (void)n_in; (void)out_size; (void)ws_size;
    const float* x      = (const float*)d_in[0];
    const float* keys   = (const float*)d_in[1];
    const float* states = (const float*)d_in[2];
    const float* Wq     = (const float*)d_in[3];
    const float* Wk     = (const float*)d_in[4];
    const float* Wv     = (const float*)d_in[5];
    const float* Wq_mha = (const float*)d_in[6];
    const float* Wk_mha = (const float*)d_in[7];
    const float* Wv_mha = (const float*)d_in[8];
    const float* Wo     = (const float*)d_in[9];
    float* out = (float*)d_out;

    const int S = 8192, D = 2048;
    char* ws = (char*)d_ws;
    const size_t SZKV = (size_t)S * D * sizeof(unsigned short);  // 32 MB
    const size_t SZW  = (size_t)D * D * sizeof(unsigned short);  // 8 MB
    unsigned short* kh  = (unsigned short*)(ws);
    unsigned short* vh  = (unsigned short*)(ws + SZKV);
    unsigned short* pe  = (unsigned short*)(ws + 2 * SZKV);
    unsigned short* wA  = (unsigned short*)(ws + 3 * SZKV);
    unsigned short* wBT = (unsigned short*)(ws + 3 * SZKV + SZW);
    unsigned short* wC  = (unsigned short*)(ws + 3 * SZKV + 2 * SZW);
    float* qv   = (float*)(ws + 3 * SZKV + 3 * SZW);
    float* qhv  = qv + D;
    float* attn = qhv + D;

    dim3 blk(256);
    dim3 gW(D / 256, D / 128);     // 2048x2048 combine GEMM
    dim3 gKV(D / 256, S / 128);    // 8192x2048 streaming GEMM

    // ---- K path: kh = rope(keys) @ (Wk_mha @ Wk)^T ----
    rope_to_bf16<<<S, blk, 0, stream>>>(keys, pe, S, D);
    convert_f32_bf16<<<(D * D) / 256, blk, 0, stream>>>(Wk_mha, wA, D * D);
    transpose_f32_bf16<<<dim3(D / 32, D / 32), blk, 0, stream>>>(Wk, wBT, D, D);
    gemm_nt_bf16<<<gW, blk, 0, stream>>>(wA, wBT, wC, D, D, D);
    gemm_nt_bf16<<<gKV, blk, 0, stream>>>(pe, wC, kh, S, D, D);

    // ---- V path: vh = rope(states) @ (Wv_mha @ Wv)^T ----
    rope_to_bf16<<<S, blk, 0, stream>>>(states, pe, S, D);
    convert_f32_bf16<<<(D * D) / 256, blk, 0, stream>>>(Wv_mha, wA, D * D);
    transpose_f32_bf16<<<dim3(D / 32, D / 32), blk, 0, stream>>>(Wv, wBT, D, D);
    gemm_nt_bf16<<<gW, blk, 0, stream>>>(wA, wBT, wC, D, D, D);
    gemm_nt_bf16<<<gKV, blk, 0, stream>>>(pe, wC, vh, S, D, D);

    // ---- q path (tiny, fp32): qh = Wq_mha @ (Wq @ x) ----
    matvec_f32<<<D / 8, blk, 0, stream>>>(Wq, x, nullptr, qv, D, D, 1.0f);
    matvec_f32<<<D / 8, blk, 0, stream>>>(Wq_mha, qv, nullptr, qhv, D, D, 1.0f);

    // ---- attention (16 heads) ----
    attn_softmax_wsum<<<16, blk, 0, stream>>>(qhv, kh, vh, attn, S);

    // ---- output: out = Wo @ attn + x ----
    matvec_f32<<<D / 8, blk, 0, stream>>>(Wo, attn, x, out, D, D, 1.0f);
}
